// Qwen3OmniVisionAttention_27711128994361
// MI455X (gfx1250) — compile-verified, matrix operand order checked by reading, not run on hardware
//
#include <hip/hip_runtime.h>

// Problem constants
#define BATCH 2
#define SEQ   2304
#define HID   1280
#define NHEAD 16
#define HDIM  80      // head dim
#define HDP   96      // head dim padded to 3x32 for K=32 WMMA
#define QKVN  3840    // 3*HID

typedef __attribute__((ext_vector_type(16))) __bf16 v16bf;
typedef __attribute__((ext_vector_type(8)))  float  v8f;

// ---------------------------------------------------------------------------
// Fragment loader for 16-bit A/B operands of V_WMMA_*_16X16X32_BF16 (wave32).
// Per CDNA5 ISA §7.12.2 (16-bit A 16x32):
//   lane: row = lane&15, g = lane>>4
//   VGPR v (0..3): K = 8g + 2v, +1 ; VGPR v (4..7): K = 16 + 8g + 2(v-4), +1
// Pairs (K, K+1) are contiguous bf16 -> u32 loads (compiler merges to b128).
// ---------------------------------------------------------------------------
__device__ __forceinline__ v16bf load_frag(const __bf16* tile, int ld) {
  const int lane = threadIdx.x & 31;
  const int g = lane >> 4;
  const __bf16* row = tile + (size_t)(lane & 15) * ld;
  v16bf f;
#pragma unroll
  for (int v = 0; v < 8; ++v) {
    const int kk = (v < 4) ? (8 * g + 2 * v) : (16 + 8 * g + 2 * (v - 4));
    union { unsigned u; __bf16 h[2]; } w;
    w.u = *(const unsigned*)(row + kk);
    f[2 * v]     = w.h[0];
    f[2 * v + 1] = w.h[1];
  }
  return f;
}

__device__ __forceinline__ v8f wmma_bf16(v16bf a, v16bf b, v8f c) {
  return __builtin_amdgcn_wmma_f32_16x16x32_bf16(
      /*neg_a=*/false, a, /*neg_b=*/false, b,
      /*c_mod=*/(short)0, c, /*reuse_a=*/false, /*reuse_b=*/false);
}

// Async global -> LDS copy (gfx1250 VGLOBAL async op, tracked by ASYNCcnt).
// VDST = LDS byte address (low 32 bits of generic shared pointer), VADDR =
// 64-bit global address, saddr = off.
__device__ __forceinline__ void async_copy_b128(unsigned lds_addr, const void* gaddr) {
  asm volatile("global_load_async_to_lds_b128 %0, %1, off"
               :: "v"(lds_addr), "v"(gaddr) : "memory");
}
__device__ __forceinline__ void wait_asynccnt0() {
  asm volatile("s_wait_asynccnt 0" ::: "memory");
}

// ---------------------------------------------------------------------------
// Stage kernels: f32 -> bf16 cast and transpose-cast for weights
// ---------------------------------------------------------------------------
__global__ void cast_bf16_kernel(const float* __restrict__ src,
                                 __bf16* __restrict__ dst, int n) {
  int i = blockIdx.x * blockDim.x + threadIdx.x;
  if (i < n) dst[i] = (__bf16)src[i];
}

// src is [K][N] row-major f32; dst is [N][K] bf16
__global__ void transpose_cast_kernel(const float* __restrict__ src,
                                      __bf16* __restrict__ dst, int K, int N) {
  int i = blockIdx.x * blockDim.x + threadIdx.x;
  if (i >= K * N) return;
  int k = i % K;
  int n = i / K;
  dst[(size_t)n * K + k] = (__bf16)src[(size_t)k * N + n];
}

// ---------------------------------------------------------------------------
// Generic bf16 WMMA GEMM: C = A[MxK] * B^T[NxK] + bias, one wave per 16x64.
// outBf != nullptr -> bf16 output, else f32 output.
// ---------------------------------------------------------------------------
__global__ __launch_bounds__(256)
void gemm_bf16_kernel(const __bf16* __restrict__ A, const __bf16* __restrict__ BT,
                      const float* __restrict__ bias,
                      __bf16* __restrict__ outBf, float* __restrict__ outF,
                      int M, int N, int K) {
  const int wid  = (blockIdx.x * blockDim.x + threadIdx.x) >> 5;
  const int lane = threadIdx.x & 31;
  const int nt4  = N >> 6;                // number of 64-wide column tiles
  const int m0   = (wid / nt4) << 4;
  const int n0   = (wid % nt4) << 6;
  if (m0 >= M) return;

  v8f acc[4] = {};
  for (int k0 = 0; k0 < K; k0 += 32) {
    v16bf a = load_frag(A + (size_t)m0 * K + k0, K);
#pragma unroll
    for (int t = 0; t < 4; ++t) {
      v16bf b = load_frag(BT + (size_t)(n0 + 16 * t) * K + k0, K);
      acc[t] = wmma_bf16(a, b, acc[t]);
    }
  }

  const int g = lane >> 4;
#pragma unroll
  for (int t = 0; t < 4; ++t) {
    const int n = n0 + 16 * t + (lane & 15);
    const float bv = bias ? bias[n] : 0.0f;
#pragma unroll
    for (int r = 0; r < 8; ++r) {
      const int m = m0 + r + 8 * g;
      const float v = acc[t][r] + bv;
      if (outBf) outBf[(size_t)m * N + n] = (__bf16)v;
      else       outF [(size_t)m * N + n] = v;
    }
  }
}

// ---------------------------------------------------------------------------
// RoPE + repack: qkv[B*S][3840] bf16 ->
//   Qp,Kp : [B][NH][S][96] bf16 (RoPE'd, Q pre-scaled by HD^-0.5, zero pad)
//   vT    : [B][NH][80][S] bf16 (transposed so P*V B-frags load contiguously)
// ---------------------------------------------------------------------------
__global__ void rope_pack_kernel(const __bf16* __restrict__ qkv,
                                 const float* __restrict__ cosb,
                                 const float* __restrict__ sinb,
                                 __bf16* __restrict__ Qp, __bf16* __restrict__ Kp,
                                 __bf16* __restrict__ vT) {
  int idx = blockIdx.x * blockDim.x + threadIdx.x;
  const int total = BATCH * NHEAD * SEQ * HDP;
  if (idx >= total) return;
  const int d = idx % HDP;
  const int s = (idx / HDP) % SEQ;
  const int h = (idx / (HDP * SEQ)) % NHEAD;
  const int b = idx / (HDP * SEQ * NHEAD);

  if (d >= HDIM) { Qp[idx] = (__bf16)0.0f; Kp[idx] = (__bf16)0.0f; return; }

  const size_t row = (size_t)(b * SEQ + s) * QKVN;
  const int f = h * HDIM + d;
  const float qv = (float)qkv[row + f];
  const float kv = (float)qkv[row + HID + f];
  const float vv = (float)qkv[row + 2 * HID + f];

  const float c  = cosb[s * HDIM + d];
  const float sn = sinb[s * HDIM + d];
  const int   dp  = (d < HDIM / 2) ? d + HDIM / 2 : d - HDIM / 2;
  const float sgn = (d < HDIM / 2) ? -1.0f : 1.0f;
  const float qr = sgn * (float)qkv[row + h * HDIM + dp];
  const float kr = sgn * (float)qkv[row + HID + h * HDIM + dp];

  const float scaling = 0.11180339887498949f;  // 80^-0.5
  Qp[idx] = (__bf16)((qv * c + qr * sn) * scaling);
  Kp[idx] = (__bf16)(kv * c + kr * sn);
  vT[((size_t)(b * NHEAD + h) * HDIM + d) * SEQ + s] = (__bf16)vv;
}

// ---------------------------------------------------------------------------
// Block-cooperative flash attention.
// One block (8 waves) = 8 consecutive 16-query tiles of the SAME (b,h).
// Per 32-key chunk: K tile (32x96 bf16) and V tile (80x32 bf16) are staged
// into LDS once per block via global_load_async_to_lds_b128 (ASYNCcnt), then
// all 8 waves run 6 WMMA (Q.K^T) + online softmax + 5 WMMA (P.V) off LDS.
// ---------------------------------------------------------------------------
__global__ __launch_bounds__(256)
void attn_kernel(const __bf16* __restrict__ Qp, const __bf16* __restrict__ Kp,
                 const __bf16* __restrict__ vT, __bf16* __restrict__ O) {
  __shared__ __align__(16) __bf16 sK[32 * HDP];     // key-major, ld=96 (6 KB)
  __shared__ __align__(16) __bf16 sV[HDIM * 32];    // dim-major, ld=32 (5 KB)
  __shared__ __align__(16) __bf16 sP[8][16 * 32];   // per-wave P tiles (8 KB)

  const int tid   = threadIdx.x;
  const int wslot = tid >> 5;
  const int lane  = tid & 31;
  const int g     = lane >> 4;

  const int QT  = SEQ / 16;                 // 144 query tiles per head
  const int BPH = QT / 8;                   // 18 blocks per head
  const int qt  = (blockIdx.x % BPH) * 8 + wslot;
  const int h   = (blockIdx.x / BPH) % NHEAD;
  const int b   = blockIdx.x / (BPH * NHEAD);

  const __bf16* qbase = Qp + (size_t)(b * NHEAD + h) * SEQ * HDP;
  const __bf16* kbase = Kp + (size_t)(b * NHEAD + h) * SEQ * HDP;
  const __bf16* vbase = vT + (size_t)(b * NHEAD + h) * HDIM * SEQ;

  const unsigned sK_addr = (unsigned)(unsigned long long)(const void*)&sK[0];
  const unsigned sV_addr = (unsigned)(unsigned long long)(const void*)&sV[0];

  v16bf qa[3];
#pragma unroll
  for (int i = 0; i < 3; ++i)
    qa[i] = load_frag(qbase + (size_t)(qt * 16) * HDP + 32 * i, HDP);

  v8f acc[5] = {};
  float mi[8], li[8];
#pragma unroll
  for (int r = 0; r < 8; ++r) { mi[r] = -1e30f; li[r] = 0.0f; }

  for (int j = 0; j < SEQ / 32; ++j) {
    // ---- async stage K tile: 32 rows x 96 bf16, fully contiguous 6144 B ----
    const char* kg = (const char*)(kbase + (size_t)(j * 32) * HDP);
    for (int idx = tid; idx < 384; idx += 256)               // 384 x 16 B
      async_copy_b128(sK_addr + idx * 16, kg + idx * 16);
    // ---- async stage V tile: 80 rows x 64 B (row stride SEQ*2 in global) ---
    for (int idx = tid; idx < 320; idx += 256) {             // 320 x 16 B
      const int d = idx >> 2, part = idx & 3;
      const char* vg = (const char*)(vbase + (size_t)d * SEQ + j * 32) + part * 16;
      async_copy_b128(sV_addr + d * 64 + part * 16, vg);
    }
    wait_asynccnt0();
    __syncthreads();

    // ---- scores S = Q * K^T for 32 keys (two 16x16 C tiles), B from LDS ----
    v8f s0 = {}, s1 = {};
#pragma unroll
    for (int i = 0; i < 3; ++i) {
      v16bf b0 = load_frag(&sK[0]        + 32 * i, HDP);
      v16bf b1 = load_frag(&sK[16 * HDP] + 32 * i, HDP);
      s0 = wmma_bf16(qa[i], b0, s0);
      s1 = wmma_bf16(qa[i], b1, s1);
    }

    // ---- online softmax (stats per row m = r + 8g, reduced over 16 lanes) --
    float sc[8];
#pragma unroll
    for (int r = 0; r < 8; ++r) {
      float mx = fmaxf(s0[r], s1[r]);
#pragma unroll
      for (int msk = 1; msk < 16; msk <<= 1) mx = fmaxf(mx, __shfl_xor(mx, msk, 32));
      const float mn = fmaxf(mi[r], mx);
      sc[r] = __expf(mi[r] - mn);
      const float p0 = __expf(s0[r] - mn);
      const float p1 = __expf(s1[r] - mn);
      float rs = p0 + p1;
#pragma unroll
      for (int msk = 1; msk < 16; msk <<= 1) rs += __shfl_xor(rs, msk, 32);
      li[r] = li[r] * sc[r] + rs;
      mi[r] = mn;
      // stash P into LDS: row m = r+8g, cols n and n+16 (row-major 16x32)
      sP[wslot][(r + 8 * g) * 32 + (lane & 15)]      = (__bf16)p0;
      sP[wslot][(r + 8 * g) * 32 + 16 + (lane & 15)] = (__bf16)p1;
    }

    // rescale the running output accumulator
#pragma unroll
    for (int t = 0; t < 5; ++t)
#pragma unroll
      for (int r = 0; r < 8; ++r) acc[t][r] *= sc[r];

    // cross-lane LDS dependency within the wave (DS ops are in-order per wave)
    asm volatile("s_wait_dscnt 0" ::: "memory");

    // reload P as a 16x32 bf16 A-fragment
    v16bf pa = load_frag(&sP[wslot][0], 32);

    // ---- O += P * V (V tile in LDS, dim-major ld=32) ----
#pragma unroll
    for (int t = 0; t < 5; ++t) {
      v16bf vb = load_frag(&sV[(size_t)(t * 16) * 32], 32);
      acc[t] = wmma_bf16(pa, vb, acc[t]);
    }
    __syncthreads();   // protect single-buffered sK/sV before next stage
  }

  // ---- epilogue: normalize and store into [B][S][H] bf16 for the proj GEMM
  float inv[8];
#pragma unroll
  for (int r = 0; r < 8; ++r) inv[r] = 1.0f / li[r];
#pragma unroll
  for (int t = 0; t < 5; ++t) {
    const int col = h * HDIM + t * 16 + (lane & 15);
#pragma unroll
    for (int r = 0; r < 8; ++r) {
      const int tok = qt * 16 + r + 8 * g;
      O[(size_t)(b * SEQ + tok) * HID + col] = (__bf16)(acc[t][r] * inv[r]);
    }
  }
}

// ---------------------------------------------------------------------------
// Host-side orchestration
// ---------------------------------------------------------------------------
extern "C" void kernel_launch(void* const* d_in, const int* in_sizes, int n_in,
                              void* d_out, int out_size, void* d_ws, size_t ws_size,
                              hipStream_t stream) {
  const float* hidden = (const float*)d_in[0];
  const float* cosb   = (const float*)d_in[1];
  const float* sinb   = (const float*)d_in[2];
  const float* w_qkv  = (const float*)d_in[3];
  const float* b_qkv  = (const float*)d_in[4];
  const float* w_proj = (const float*)d_in[5];
  const float* b_proj = (const float*)d_in[6];
  float* out = (float*)d_out;

  // workspace carve-up (all bf16, every section 16B-aligned)
  char* ws = (char*)d_ws;
  size_t o = 0;
  __bf16* Xbf    = (__bf16*)(ws + o); o += (size_t)BATCH * SEQ * HID * 2;
  __bf16* WqkvT  = (__bf16*)(ws + o); o += (size_t)QKVN * HID * 2;
  __bf16* WprojT = (__bf16*)(ws + o); o += (size_t)HID * HID * 2;
  __bf16* QKV    = (__bf16*)(ws + o); o += (size_t)BATCH * SEQ * QKVN * 2;
  __bf16* Qp     = (__bf16*)(ws + o); o += (size_t)BATCH * NHEAD * SEQ * HDP * 2;
  __bf16* Kp     = (__bf16*)(ws + o); o += (size_t)BATCH * NHEAD * SEQ * HDP * 2;
  __bf16* Vt     = (__bf16*)(ws + o); o += (size_t)BATCH * NHEAD * HDIM * SEQ * 2;
  __bf16* AttnO  = (__bf16*)(ws + o); o += (size_t)BATCH * SEQ * HID * 2;

  const int TB = 256;
  // 1) stage inputs to bf16 (weights transposed to [N][K])
  {
    int n = BATCH * SEQ * HID;
    cast_bf16_kernel<<<(n + TB - 1) / TB, TB, 0, stream>>>(hidden, Xbf, n);
  }
  {
    int n = HID * QKVN;
    transpose_cast_kernel<<<(n + TB - 1) / TB, TB, 0, stream>>>(w_qkv, WqkvT, HID, QKVN);
  }
  {
    int n = HID * HID;
    transpose_cast_kernel<<<(n + TB - 1) / TB, TB, 0, stream>>>(w_proj, WprojT, HID, HID);
  }

  // 2) QKV GEMM: [4608x1280] x [1280x3840] -> bf16
  {
    int M = BATCH * SEQ, N = QKVN, K = HID;
    int waves = (M / 16) * (N / 64);
    gemm_bf16_kernel<<<waves / 8, TB, 0, stream>>>(Xbf, WqkvT, b_qkv, QKV, nullptr, M, N, K);
  }

  // 3) RoPE + repack
  {
    int n = BATCH * NHEAD * SEQ * HDP;
    rope_pack_kernel<<<(n + TB - 1) / TB, TB, 0, stream>>>(QKV, cosb, sinb, Qp, Kp, Vt);
  }

  // 4) block-cooperative flash attention (8 q-tiles per block, shared K/V LDS)
  {
    int blocks = BATCH * NHEAD * ((SEQ / 16) / 8);  // 576
    attn_kernel<<<blocks, TB, 0, stream>>>(Qp, Kp, Vt, AttnO);
  }

  // 5) output projection -> f32 d_out
  {
    int M = BATCH * SEQ, N = HID, K = HID;
    int waves = (M / 16) * (N / 64);
    gemm_bf16_kernel<<<waves / 8, TB, 0, stream>>>(AttnO, WprojT, b_proj, nullptr, out, M, N, K);
  }
}
